// SelfAttention1_52948356825333
// MI455X (gfx1250) — matrix operand order repaired and checked by reading, not compiled, not run on hardware
//
#include <hip/hip_runtime.h>

// ---------------------------------------------------------------------------
// Gated multi-head self-attention (AlphaFold-style), bf16 WMMA pipeline.
// B=64, S=512, A=256, H=8, C=32.  wave32 / gfx1250.
//
// k0a: q_data/m_data f32 -> bf16 (removes cvt + halves A traffic in GEMM)
// k0b: pack 4 weight mats into exact WMMA B-fragment layout (b128 loads)
// k1 : fused projections q/k/v/gate via v_wmma_f32_16x16x32_bf16;
//      V stored transposed via LDS bounce -> one b128 store per lane
// k2 : streaming-softmax attention; K/V tiles staged to LDS with
//      global_load_async_to_lds_b128 double buffering (ASYNCcnt pipelined)
// ---------------------------------------------------------------------------

typedef __attribute__((ext_vector_type(16))) __bf16 v16bf;
typedef __attribute__((ext_vector_type(8)))  __bf16 v8bf;
typedef __attribute__((ext_vector_type(4)))  __bf16 v4bf;
typedef __attribute__((ext_vector_type(8)))  float  v8f;

#define BDIM 64
#define SDIM 512
#define ADIM 256
#define HDIM 8
#define CDIM 32
#define NCOL 256   // H*C
#define WPACK_PER_MAT (16 * 8 * 32 * 16)   // ntile * ktile * lane * elem = 65536

static __device__ __forceinline__ __bf16 to_bf16(float f) { return (__bf16)f; }

static __device__ __forceinline__ v8f wmma_bf16(v16bf a, v16bf b, v8f c) {
  return __builtin_amdgcn_wmma_f32_16x16x32_bf16(false, a, false, b, (short)0, c,
                                                 false, false);
}

// 32B fragment from two 16B-aligned halves (LDS rows are 80B-strided).
static __device__ __forceinline__ v16bf load16(const __bf16* p) {
  const v8bf lo = *(const v8bf*)p;
  const v8bf hi = *(const v8bf*)(p + 8);
  return __builtin_shufflevector(lo, hi, 0, 1, 2, 3, 4, 5, 6, 7,
                                 8, 9, 10, 11, 12, 13, 14, 15);
}
// A-fragment: two 8-element runs (K 16*hl+0..7 / 16*hl+8..15).
static __device__ __forceinline__ v16bf load8x2(const __bf16* p0, const __bf16* p1) {
  const v8bf lo = *(const v8bf*)p0;
  const v8bf hi = *(const v8bf*)p1;
  return __builtin_shufflevector(lo, hi, 0, 1, 2, 3, 4, 5, 6, 7,
                                 8, 9, 10, 11, 12, 13, 14, 15);
}

// Async copy of one contiguous 64B row into LDS (4 x b128; the same IOFFSET
// is applied to both LDS and global address per ISA 15.18.3 semantics).
static __device__ __forceinline__ void async_copy_row64(const __bf16* gsrc,
                                                        __bf16* ldst) {
  const unsigned long long g = (unsigned long long)(uintptr_t)gsrc;
  const unsigned int l = (unsigned int)(uintptr_t)ldst;  // flat[31:0] = LDS offset
  asm volatile(
      "global_load_async_to_lds_b128 %0, %1, off\n\t"
      "global_load_async_to_lds_b128 %0, %1, off offset:16\n\t"
      "global_load_async_to_lds_b128 %0, %1, off offset:32\n\t"
      "global_load_async_to_lds_b128 %0, %1, off offset:48"
      :: "v"(l), "v"(g) : "memory");
}

// ---------------------------------------------------------------------------
// Kernel 0a: f32 -> bf16 bulk convert (4 elements / thread).
// ---------------------------------------------------------------------------
__global__ __launch_bounds__(256)
void af_cvt_kernel(const float* __restrict__ src, __bf16* __restrict__ dst) {
  const int i = blockIdx.x * 256 + threadIdx.x;        // one float4 per thread
  const float4 f = ((const float4*)src)[i];
  v4bf o;
  o[0] = to_bf16(f.x); o[1] = to_bf16(f.y);
  o[2] = to_bf16(f.z); o[3] = to_bf16(f.w);
  ((v4bf*)dst)[i] = o;
}

// ---------------------------------------------------------------------------
// Kernel 0b: pack weights [A=256, HC=256] f32 into bf16 B-fragment layout:
//   wpack[mat][ntile][kt][lane][i] = w[(kt*32 + 16*(lane>>4) + i)*256
//                                      + ntile*16 + (lane&15)]
// so a wave's B-fragment is one contiguous 32B chunk per lane.
// ---------------------------------------------------------------------------
__global__ __launch_bounds__(256)
void af_pack_w_kernel(const float* __restrict__ qw, const float* __restrict__ kw,
                      const float* __restrict__ vw, const float* __restrict__ gw,
                      __bf16* __restrict__ wpack) {
  const int gid = blockIdx.x * 256 + threadIdx.x;      // 0 .. 4*65536-1
  const int i     = gid & 15;
  const int lane  = (gid >> 4) & 31;
  const int kt    = (gid >> 9) & 7;
  const int ntile = (gid >> 12) & 15;
  const int mat   = gid >> 16;
  const float* w = (mat == 0) ? qw : (mat == 1) ? kw : (mat == 2) ? vw : gw;
  const int a   = kt * 32 + 16 * (lane >> 4) + i;
  const int col = ntile * 16 + (lane & 15);
  wpack[gid] = to_bf16(w[(size_t)a * NCOL + col]);
}

// ---------------------------------------------------------------------------
// Kernel 1: fused projections (one 16x16 tile x 4 matrices per wave).
//   q  = (q_data @ query_w + query_b) * C^-0.5  -> bf16 [b,s,h,c]
//   k  =  m_data @ key_w                        -> bf16 [b,s,h,c]
//   v  =  m_data @ value_w                      -> bf16 [b,h,c,s] (transposed)
//   g  =  sigmoid(q_data @ gating_w)            -> f32  [b,s,h,c]
// ---------------------------------------------------------------------------
__global__ __launch_bounds__(128)
void af_proj_kernel(const __bf16* __restrict__ qdb, const __bf16* __restrict__ mdb,
                    const __bf16* __restrict__ wpack, const float* __restrict__ qb,
                    __bf16* __restrict__ q_bf, __bf16* __restrict__ k_bf,
                    __bf16* __restrict__ vt_bf, float* __restrict__ gate)
{
  __shared__ __bf16 vls[4][16][24];             // per-wave V-transpose tile

  const int lane = threadIdx.x & 31;
  const int wave = threadIdx.x >> 5;
  const int task = blockIdx.x * 4 + wave;       // 32768 wave-tasks
  const int mtile = task >> 4;                  // 16-row tiles of B*S
  const int ntile = task & 15;                  // 16-col tiles of H*C
  const int n  = lane & 15;
  const int hl = lane >> 4;
  const int col  = ntile * 16 + n;
  const int row0 = mtile * 16;

  v8f accq = {}, acck = {}, accv = {}, accg = {};

  const __bf16* qrow = qdb + (size_t)(row0 + n) * ADIM;   // A rows: m = lane&15
  const __bf16* mrow = mdb + (size_t)(row0 + n) * ADIM;
  const __bf16* wq = wpack + 0 * WPACK_PER_MAT + ((size_t)ntile * 8) * 32 * 16;
  const __bf16* wk = wq + 1 * WPACK_PER_MAT;
  const __bf16* wv = wq + 2 * WPACK_PER_MAT;
  const __bf16* wg = wq + 3 * WPACK_PER_MAT;

#pragma unroll
  for (int kt = 0; kt < 8; ++kt) {
    const int a0 = kt * 32 + 16 * hl;           // A: elems 0..7
    const int a1 = kt * 32 + 16 * hl + 8;       // A: elems 8..15
    const v16bf aq = load8x2(qrow + a0, qrow + a1);
    const v16bf am = load8x2(mrow + a0, mrow + a1);
    const size_t wof = ((size_t)kt * 32 + lane) * 16;   // one 32B chunk / lane
    const v16bf bq = *(const v16bf*)(wq + wof);
    const v16bf bk = *(const v16bf*)(wk + wof);
    const v16bf bv = *(const v16bf*)(wv + wof);
    const v16bf bg = *(const v16bf*)(wg + wof);
    accq = wmma_bf16(aq, bq, accq);
    acck = wmma_bf16(am, bk, acck);
    accv = wmma_bf16(am, bv, accv);
    accg = wmma_bf16(aq, bg, accg);
  }

  const float qbias = qb[col];
  const float scale = 0.17677669529663687f;     // C^-0.5
#pragma unroll
  for (int j = 0; j < 8; ++j) {                 // D layout: row = j + 8*hl
    const int r = row0 + j + 8 * hl;            // global row = b*S + s
    const size_t o = (size_t)r * NCOL + col;
    q_bf[o] = to_bf16((accq[j] + qbias) * scale);   // 16-lane contiguous
    k_bf[o] = to_bf16(acck[j]);
    gate[o] = 1.0f / (1.0f + __expf(-accg[j]));
    vls[wave][n][j + 8 * hl] = to_bf16(accv[j]);    // transpose V via LDS
  }
  asm volatile("s_wait_dscnt 0" ::: "memory");      // intra-wave LDS RAW fence

  // V^T store: lane owns (c_local = lane&15, s-half = lane>>4) -> one b128.
  {
    const int h = (ntile * 16) >> 5;            // wave-uniform head
    const int cbase = (ntile & 1) * 16;         // wave-uniform c base
    const int b = row0 >> 9, s0 = row0 & 511;
    const v8bf vrow = *(const v8bf*)(&vls[wave][n][8 * hl]);
    __bf16* dst = vt_bf + (((size_t)b * HDIM + h) * CDIM + cbase + n) * SDIM
                        + s0 + 8 * hl;
    *(v8bf*)dst = vrow;                         // 16 contiguous bytes of s
  }
}

// ---------------------------------------------------------------------------
// Kernel 2: streaming softmax attention, one wave per (b, h, 16-row q tile).
// K/V 32-key tiles double-buffered in LDS via async b128 copies (ASYNCcnt).
// ---------------------------------------------------------------------------
__global__ __launch_bounds__(128)
void af_attn_kernel(const float* __restrict__ bias, const float* __restrict__ nbias,
                    const __bf16* __restrict__ q_bf, const __bf16* __restrict__ k_bf,
                    const __bf16* __restrict__ vt_bf, const float* __restrict__ gate,
                    float* __restrict__ out)
{
  // 80B row stride => 16 lanes touch 16 distinct banks on fragment reads.
  __shared__ __bf16 kbuf[4][2][32][40];
  __shared__ __bf16 vbuf[4][2][32][40];
  __shared__ __bf16 pls[4][16][48];             // per-wave P-transpose tile

  const int lane = threadIdx.x & 31;
  const int wave = threadIdx.x >> 5;
  const int task = blockIdx.x * 4 + wave;       // 16384 wave-tasks
  const int qt = task & 31;
  const int h  = (task >> 5) & 7;
  const int b  = task >> 8;
  const int n  = lane & 15;
  const int hl = lane >> 4;
  const int q0 = qt * 16;

  // Q A-fragment held in registers for the whole key loop.
  const __bf16* qrow = q_bf + ((size_t)(b * SDIM + q0 + n) * NCOL + h * CDIM);
  const v16bf qfrag = load8x2(qrow + 16 * hl, qrow + 16 * hl + 8);

  v8f acc0 = {}, acc1 = {};
  float mrun[8], lrun[8];
#pragma unroll
  for (int j = 0; j < 8; ++j) { mrun[j] = -1e30f; lrun[j] = 0.0f; }

  const float* brow = bias + (size_t)b * SDIM * SDIM;
  const float* nrow = nbias + (size_t)h * SDIM * SDIM;
  const __bf16* kstage = k_bf + (size_t)b * SDIM * NCOL + h * CDIM;  // [key][c]
  const __bf16* vtb = vt_bf + ((size_t)b * HDIM + h) * CDIM * SDIM;  // [c][key]

  // ---- async stage of one 32-key tile: lane copies one 64B row of K and V --
  auto issue_tile = [&](int buf, int key0) {
    async_copy_row64(kstage + (size_t)(key0 + lane) * NCOL, &kbuf[wave][buf][lane][0]);
    async_copy_row64(vtb + (size_t)lane * SDIM + key0,      &vbuf[wave][buf][lane][0]);
  };

  issue_tile(0, 0);                             // prologue prefetch

  for (int key0 = 0; key0 < SDIM; key0 += 32) {
    const int cur = (key0 >> 5) & 1;
    if (key0 + 32 < SDIM) {
      issue_tile(cur ^ 1, key0 + 32);           // keep next tile in flight
      asm volatile("s_wait_asynccnt 8" ::: "memory");   // drain previous tile
    } else {
      asm volatile("s_wait_asynccnt 0" ::: "memory");
    }

    // ---- logits = Q @ K^T (contraction over C=32; B elem i -> c=16*hl+i) ---
    const v16bf kb0 = load16(&kbuf[wave][cur][n][16 * hl]);
    const v16bf kb1 = load16(&kbuf[wave][cur][n + 16][16 * hl]);
    v8f lg0 = {}, lg1 = {};
    lg0 = wmma_bf16(qfrag, kb0, lg0);
    lg1 = wmma_bf16(qfrag, kb1, lg1);

    if (key0 + 32 < SDIM)                       // stream next bias tile
      __builtin_prefetch(brow + (size_t)(q0 + hl * 8) * SDIM + key0 + 32, 0, 0);

    // ---- biases (D layout: row j+8*hl, key n / n+16) ----
    float tmax[8];
#pragma unroll
    for (int j = 0; j < 8; ++j) {
      const int qr = q0 + j + 8 * hl;
      const size_t o = (size_t)qr * SDIM + key0 + n;
      const float l0 = lg0[j] + brow[o] + nrow[o];
      const float l1 = lg1[j] + brow[o + 16] + nrow[o + 16];
      lg0[j] = l0; lg1[j] = l1;
      tmax[j] = fmaxf(l0, l1);
    }
#pragma unroll
    for (int s = 1; s < 16; s <<= 1)
#pragma unroll
      for (int j = 0; j < 8; ++j)
        tmax[j] = fmaxf(tmax[j], __shfl_xor(tmax[j], s, 32));

    // ---- online softmax update ----
    float tsum[8];
#pragma unroll
    for (int j = 0; j < 8; ++j) {
      const float nm = fmaxf(mrun[j], tmax[j]);
      const float corr = __expf(mrun[j] - nm);
      mrun[j] = nm;
      acc0[j] *= corr; acc1[j] *= corr; lrun[j] *= corr;
      const float p0 = __expf(lg0[j] - nm);
      const float p1 = __expf(lg1[j] - nm);
      lg0[j] = p0; lg1[j] = p1;
      tsum[j] = p0 + p1;
    }
#pragma unroll
    for (int s = 1; s < 16; s <<= 1)
#pragma unroll
      for (int j = 0; j < 8; ++j)
        tsum[j] += __shfl_xor(tsum[j], s, 32);
#pragma unroll
    for (int j = 0; j < 8; ++j) lrun[j] += tsum[j];

    // ---- P: C/D layout -> A layout via per-wave LDS bounce ----
#pragma unroll
    for (int j = 0; j < 8; ++j) {
      pls[wave][j + 8 * hl][n]      = to_bf16(lg0[j]);
      pls[wave][j + 8 * hl][n + 16] = to_bf16(lg1[j]);
    }
    asm volatile("s_wait_dscnt 0" ::: "memory");   // intra-wave LDS RAW fence
    const __bf16* prow = &pls[wave][n][0];
    const v16bf pfrag = load8x2(prow + 16 * hl, prow + 16 * hl + 8);

    // ---- acc += P @ V (V rows in LDS are contiguous over key) ----
    const v16bf vb0 = load16(&vbuf[wave][cur][n][16 * hl]);
    const v16bf vb1 = load16(&vbuf[wave][cur][n + 16][16 * hl]);
    acc0 = wmma_bf16(pfrag, vb0, acc0);
    acc1 = wmma_bf16(pfrag, vb1, acc1);
  }

  // ---- epilogue: /softmax-denominator, apply sigmoid gate, NT stores ----
#pragma unroll
  for (int j = 0; j < 8; ++j) {
    const int qr = q0 + j + 8 * hl;
    const float inv = 1.0f / lrun[j];
    const size_t o = (size_t)(b * SDIM + qr) * NCOL + h * CDIM;
    __builtin_nontemporal_store(acc0[j] * inv * gate[o + n],      &out[o + n]);
    __builtin_nontemporal_store(acc1[j] * inv * gate[o + 16 + n], &out[o + 16 + n]);
  }
}

// ---------------------------------------------------------------------------
extern "C" void kernel_launch(void* const* d_in, const int* in_sizes, int n_in,
                              void* d_out, int out_size, void* d_ws, size_t ws_size,
                              hipStream_t stream) {
  const float* q_data = (const float*)d_in[0];
  const float* m_data = (const float*)d_in[1];
  const float* bias   = (const float*)d_in[2];
  const float* nbias  = (const float*)d_in[3];
  const float* qw     = (const float*)d_in[4];
  const float* qb     = (const float*)d_in[5];
  const float* kw     = (const float*)d_in[6];
  const float* vw     = (const float*)d_in[7];
  const float* gw     = (const float*)d_in[8];
  float* out = (float*)d_out;

  char* ws = (char*)d_ws;
  const size_t e = (size_t)BDIM * SDIM * NCOL;       // 8,388,608 elements
  __bf16* q_bf  = (__bf16*)(ws);                     // 16 MB
  __bf16* k_bf  = (__bf16*)(ws + 2 * e);             // 16 MB
  __bf16* vt_bf = (__bf16*)(ws + 4 * e);             // 16 MB (transposed V)
  float*  gate  = (float*)(ws + 6 * e);              // 32 MB
  __bf16* qd_bf = (__bf16*)(ws + 10 * e);            // 16 MB
  __bf16* md_bf = (__bf16*)(ws + 12 * e);            // 16 MB
  __bf16* wpack = (__bf16*)(ws + 14 * e);            // 512 KB

  af_cvt_kernel<<<8192, 256, 0, stream>>>(q_data, qd_bf);   // e/4 float4's
  af_cvt_kernel<<<8192, 256, 0, stream>>>(m_data, md_bf);
  af_pack_w_kernel<<<1024, 256, 0, stream>>>(qw, kw, vw, gw, wpack);
  af_proj_kernel<<<8192, 128, 0, stream>>>(qd_bf, md_bf, wpack, qb,
                                           q_bf, k_bf, vt_bf, gate);
  af_attn_kernel<<<4096, 128, 0, stream>>>(bias, nbias, q_bf, k_bf, vt_bf, gate,
                                           out);
}
